// DimeNet_60198261620809
// MI455X (gfx1250) — matrix operand order repaired and checked by reading, not compiled, not run on hardware
//
#include <hip/hip_runtime.h>
#include <math.h>

// ---------------------------------------------------------------------------
// DimeNet forward on gfx1250 (MI455X). GEMMs: V_WMMA_F32_16X16X32_BF16
// (bf16 in, fp32 accumulate). Weight panels staged to LDS via the Tensor
// Data Mover (tensor_load_to_lds + s_wait_tensorcnt) when available.
// ---------------------------------------------------------------------------
#define N_ATOMS 8000
#define N_EDGES 100000
#define N_TRIP  500000
#define N_GRAPH 64
#define HD      128          // H
#define NR      6
#define NSR     42           // NS*NR
#define NB      8
#define NBLK    6
#define KTRIP   (NB*HD)      // 1024 folded K for triplet einsum

typedef __attribute__((ext_vector_type(16))) __bf16 bf16x16;
typedef __attribute__((ext_vector_type(8)))  float  f32x8;
typedef __attribute__((ext_vector_type(4)))  unsigned int u32x4;
typedef __attribute__((ext_vector_type(8)))  int  i32x8;
typedef __attribute__((ext_vector_type(4)))  int  i32x4;

#define STR2(x) #x
#define STR(x) STR2(x)
#pragma message("CDNA5 build: clang major = " STR(__clang_major__))

#if defined(__has_builtin)
#if __has_builtin(__builtin_amdgcn_tensor_load_to_lds) && \
    __has_builtin(__builtin_amdgcn_s_wait_tensorcnt)
#define USE_TDM 1
#endif
#endif

#if defined(USE_TDM)
#pragma message("CDNA5: TDM path ENABLED (tensor_load_to_lds + s_wait_tensorcnt)")
#else
#pragma message("CDNA5: TDM builtin NOT found - cooperative-copy fallback in use")
#endif

__device__ __forceinline__ float siluf(float v) {
    // v * rcp(1+exp(-v)) : v_exp_f32 + v_rcp_f32, no IEEE-divide expansion
    return v * __builtin_amdgcn_rcpf(1.0f + __expf(-v));
}
__device__ __forceinline__ unsigned short f2bf(float f) {
    union { float f; unsigned u; } c; c.f = f;
    unsigned u = c.u + 0x7FFFu + ((c.u >> 16) & 1u);   // round-to-nearest-even
    return (unsigned short)(u >> 16);
}
__device__ __forceinline__ float bf2f(unsigned short h) {
    union { unsigned u; float f; } c; c.u = ((unsigned)h) << 16; return c.f;
}

// ---------------------------------------------------------------------------
// TDM: 1-D DMA of `nelem` bf16 elements from global -> LDS. Issued by one
// wave; caller must s_wait_tensorcnt + barrier before consuming.
// D# packing per CDNA5 ISA 8.3/8.4 (group0: count|lds|global|type=2,
// group1: data_size=2B, tensor_dim0 = tile_dim0 = nelem, 1-D tile).
// ---------------------------------------------------------------------------
__device__ __forceinline__ void tdm_load_1d(void* lds_dst, const void* gsrc,
                                            unsigned nelem /* <= 65535 */) {
    unsigned lds_off = (unsigned)(unsigned long long)(uintptr_t)lds_dst; // low 32b = LDS offset
    unsigned long long ga = (unsigned long long)(uintptr_t)gsrc;
    u32x4 g0;
    g0[0] = 1u;                                            // count=1, user D#
    g0[1] = lds_off;                                       // lds_addr
    g0[2] = (unsigned)(ga & 0xFFFFFFFFu);                  // global_addr[31:0]
    g0[3] = (unsigned)((ga >> 32) & 0x01FFFFFFu) | (2u << 30); // ga[56:32] | type=2
    i32x8 g1;
    g1[0] = 0x00010000;                                    // data_size=1 (2B), mask=0
    g1[1] = (int)((nelem & 0xFFFFu) << 16);                // tensor_dim0[15:0]
    g1[2] = (int)((nelem >> 16) & 0xFFFFu);                // tensor_dim0[31:16], td1=0
    g1[3] = (int)((nelem & 0xFFFFu) << 16);                // tile_dim0
    g1[4] = 0;                                             // tile_dim1=0, tile_dim2=0
    g1[5] = (int)nelem;                                    // tensor_dim0_stride[31:0]
    g1[6] = 0;
    g1[7] = 0;
#if defined(USE_TDM)
    i32x4 z4 = {};
#if __clang_major__ >= 23
    i32x8 z8 = {};
    __builtin_amdgcn_tensor_load_to_lds(g0, g1, z4, z4, z8, 0);
#else
    __builtin_amdgcn_tensor_load_to_lds(g0, g1, z4, z4, 0);
#endif
#endif
}

// Stage a bf16 weight panel of `nelem` elements into LDS. Wave 0 drives the
// TDM engine (TENSORcnt); other waves just hit the barrier.
__device__ __forceinline__ void stage_weights(unsigned short* lds_dst,
                                              const unsigned short* gsrc,
                                              unsigned nelem) {
#if defined(USE_TDM)
    if (threadIdx.x < 32) {
        tdm_load_1d(lds_dst, gsrc, nelem);
        __builtin_amdgcn_s_wait_tensorcnt(0);
    }
#else
    for (unsigned t = threadIdx.x * 8u; t < nelem; t += blockDim.x * 8u)
        *(uint4*)(lds_dst + t) = *(const uint4*)(gsrc + t);
#endif
    __syncthreads();
}

// ---------------------------------------------------------------------------
// Wave-level 16x16 WMMA accumulate: D += A(16xK bf16 rows) x B(Kx16).
// A layout: lane m/half h holds the contiguous run [k0+16h,+16).
// B fragment: 16 consecutive K of column n -> Bt[n][K]
// (transposed) so it's one contiguous 32B load.
// ---------------------------------------------------------------------------
__device__ __forceinline__ void wmma_accum(const unsigned short* __restrict__ rowbase,
                                           int ldk,
                                           const unsigned short* __restrict__ Bt,
                                           int K, int col0, f32x8& acc) {
    const int lane = threadIdx.x & 31;
    const int half = lane >> 4;
    const int m    = lane & 15;
    const unsigned short* arow = rowbase + m * ldk;
    const unsigned short* bcol = Bt + (size_t)(col0 + m) * K;   // m == n for B
    for (int k0 = 0; k0 < K; k0 += 32) {
        bf16x16 a, b;
        __builtin_memcpy(&a,             arow + k0 + half * 16,     32);
        __builtin_memcpy(&b,             bcol + k0 + half * 16,     32);
        acc = __builtin_amdgcn_wmma_f32_16x16x32_bf16(false, a, false, b,
                                                      (short)0, acc, false, false);
    }
}

// Same but two adjacent 16-col tiles share one A fragment (2 WMMA per K-step).
__device__ __forceinline__ void wmma_accum2(const unsigned short* __restrict__ rowbase,
                                            int ldk,
                                            const unsigned short* __restrict__ Bt,
                                            int K, int cbase,
                                            f32x8& acc0, f32x8& acc1) {
    const int lane = threadIdx.x & 31;
    const int half = lane >> 4;
    const int m    = lane & 15;
    const unsigned short* arow  = rowbase + m * ldk;
    const unsigned short* bcol0 = Bt + (size_t)(cbase + m) * K;
    const unsigned short* bcol1 = Bt + (size_t)(cbase + 16 + m) * K;
    for (int k0 = 0; k0 < K; k0 += 32) {
        bf16x16 a, b0, b1;
        __builtin_memcpy(&a,             arow + k0 + half * 16,     32);
        __builtin_memcpy(&b0,            bcol0 + k0 + half * 16,    32);
        __builtin_memcpy(&b1,            bcol1 + k0 + half * 16,    32);
        acc0 = __builtin_amdgcn_wmma_f32_16x16x32_bf16(false, a, false, b0,
                                                       (short)0, acc0, false, false);
        acc1 = __builtin_amdgcn_wmma_f32_16x16x32_bf16(false, a, false, b1,
                                                       (short)0, acc1, false, false);
    }
}

// ---------------------------------------------------------------------------
// Weight conversion kernels (run once per launch; tiny).
// ---------------------------------------------------------------------------
__global__ void k_conv(const float* __restrict__ s, unsigned short* __restrict__ d, int n) {
    int i = blockIdx.x * blockDim.x + threadIdx.x;
    if (i < n) d[i] = f2bf(s[i]);
}
// src [K,N] fp32 row-major -> dst [N,K] bf16 (transposed)
__global__ void k_convT(const float* __restrict__ s, unsigned short* __restrict__ d,
                        int K, int N) {
    int i = blockIdx.x * blockDim.x + threadIdx.x;
    if (i >= K * N) return;
    int k = i / N, n = i - k * N;
    d[(size_t)n * K + k] = f2bf(s[i]);
}
// int_W[b]: src [l(128)][j(8)][i(128)] -> dst[i][j*128+l]  (B^T for folded einsum)
__global__ void k_convW(const float* __restrict__ s, unsigned short* __restrict__ d) {
    int g = blockIdx.x * blockDim.x + threadIdx.x;
    if (g >= 128 * 8 * 128) return;
    int l = g >> 10, jj = (g >> 7) & 7, i = g & 127;
    d[(size_t)i * KTRIP + jj * 128 + l] = f2bf(s[g]);
}

__global__ void k_zerof(float* p, int n) {
    int i = blockIdx.x * blockDim.x + threadIdx.x;
    if (i < n) p[i] = 0.0f;
}

// ---------------------------------------------------------------------------
// Generic fused MLP layer: Y = epilogue(X[R,128] @ W + bias)
// 512 threads = 16 waves; tile 64 rows x 128 cols; wave = 16 rows x 32 cols
// (two col tiles share one A fragment -> 8 WMMA/wave). B panel (32KB) via TDM.
// Epilogue: +bias -> (SiLU) -> (* rbf@rbfw) -> (+resid) -> write f32 / bf16.
// ---------------------------------------------------------------------------
__global__ __launch_bounds__(512)
void k_mlp(const float* __restrict__ X, const unsigned short* __restrict__ Wt,
           const float* __restrict__ bias, int R,
           float* __restrict__ Yf, unsigned short* __restrict__ Ybf,
           const float* __restrict__ resid, int do_act,
           const float* __restrict__ rbf6, const float* __restrict__ rbfw) {
    __shared__ unsigned short Abuf[64 * HD];        // 16KB activation tile
    __shared__ unsigned short Bbuf[HD * HD];        // 32KB weight panel
    const int row0 = blockIdx.x * 64;
    for (int t = threadIdx.x; t < 64 * HD; t += 512) {
        int r = t >> 7, c = t & 127;
        float v = (row0 + r < R) ? X[(size_t)(row0 + r) * HD + c] : 0.0f;
        Abuf[t] = f2bf(v);
    }
    stage_weights(Bbuf, Wt, HD * HD);               // includes __syncthreads()
    const int wave = threadIdx.x >> 5, lane = threadIdx.x & 31;
    const int rtile = (wave & 3) * 16;              // 4 row tiles
    const int cbase = (wave >> 2) * 32;             // 4 col groups of 32
    f32x8 acc0 = {}, acc1 = {};
    wmma_accum2(Abuf + rtile * HD, HD, Bbuf, HD, cbase, acc0, acc1);
    const int half = lane >> 4, n = lane & 15;
    #pragma unroll
    for (int r = 0; r < 8; ++r) {
        int row = row0 + rtile + r + 8 * half;
        if (row >= R) continue;
        #pragma unroll
        for (int cc = 0; cc < 2; ++cc) {
            int col = cbase + cc * 16 + n;
            float y = (cc ? acc1[r] : acc0[r]) + bias[col];
            if (do_act) y = siluf(y);
            if (rbfw) {                  // y *= (rbf[row,:6] @ rbfw[:, col])
                float rp = 0.0f;
                #pragma unroll
                for (int s = 0; s < NR; ++s)
                    rp += rbf6[(size_t)row * NR + s] * rbfw[s * HD + col];
                y *= rp;
            }
            if (resid) y += resid[(size_t)row * HD + col];
            if (Yf)  Yf[(size_t)row * HD + col] = y;
            if (Ybf) Ybf[(size_t)row * HD + col] = f2bf(y);
        }
    }
}

// ---------------------------------------------------------------------------
// Embedding: x = silu(concat(h[i], h[j], silu(rbf@Wr+br)) @ emb_w + emb_b)
// K = 384; A built in LDS (bf16), 32 edges per block; 96KB weight panel via TDM.
// ---------------------------------------------------------------------------
__global__ __launch_bounds__(512)
void k_embed(const float* __restrict__ rbf, const int* __restrict__ iidx,
             const int* __restrict__ jidx, const int* __restrict__ z,
             const unsigned short* __restrict__ emb_bf,
             const float* __restrict__ emb_rbf_w, const float* __restrict__ emb_rbf_b,
             const unsigned short* __restrict__ embWt, const float* __restrict__ emb_b,
             float* __restrict__ X, int R) {
    __shared__ unsigned short Abuf[32 * 384];
    __shared__ unsigned short Bbuf[HD * 384];       // 96KB
    const int row0 = blockIdx.x * 32;
    const int tid = threadIdx.x, ee = tid >> 4, sub = tid & 15;
    const int e = row0 + ee;
    int zi = 0, zj = 0;
    if (e < R) { zi = z[iidx[e]]; zj = z[jidx[e]]; }
    #pragma unroll 4
    for (int q = 0; q < 24; ++q) {
        int c = sub + q * 16;                      // 0..383
        float v = 0.0f;
        if (e < R) {
            if (c < 128)       v = bf2f(emb_bf[(size_t)zi * HD + c]);
            else if (c < 256)  v = bf2f(emb_bf[(size_t)zj * HD + (c - 128)]);
            else {
                int cc = c - 256;
                float s = emb_rbf_b[cc];
                #pragma unroll
                for (int r6 = 0; r6 < NR; ++r6)
                    s += rbf[(size_t)e * NR + r6] * emb_rbf_w[r6 * HD + cc];
                v = siluf(s);
            }
        }
        Abuf[ee * 384 + c] = f2bf(v);
    }
    stage_weights(Bbuf, embWt, HD * 384);           // includes __syncthreads()
    const int wave = tid >> 5, lane = tid & 31;
    const int rtile = (wave & 1) * 16, col0 = (wave >> 1) * 16;
    f32x8 acc = {};
    wmma_accum(Abuf + rtile * 384, 384, Bbuf, 384, col0, acc);
    const int half = lane >> 4, col = col0 + (lane & 15);
    #pragma unroll
    for (int r = 0; r < 8; ++r) {
        int row = row0 + rtile + r + 8 * half;
        if (row < R) X[(size_t)row * HD + col] = siluf(acc[r] + emb_b[col]);
    }
}

// ---------------------------------------------------------------------------
// sbf_p = sbf[T,42] @ w[42,8]  (memory-bound, plain VALU)
// ---------------------------------------------------------------------------
__global__ void k_sbfp(const float* __restrict__ sbf, const float* __restrict__ w,
                       float* __restrict__ out, int T) {
    int t = blockIdx.x * blockDim.x + threadIdx.x;
    if (t >= T) return;
    float a[NB] = {};
    #pragma unroll 6
    for (int s = 0; s < NSR; ++s) {
        float sv = sbf[(size_t)t * NSR + s];
        #pragma unroll
        for (int jj = 0; jj < NB; ++jj) a[jj] += sv * w[s * NB + jj];
    }
    #pragma unroll
    for (int jj = 0; jj < NB; ++jj) out[(size_t)t * NB + jj] = a[jj];
}

// ---------------------------------------------------------------------------
// Triplet einsum, folded: A'[t, j*128+l] = sbf_p[t,j] * x_kj[idx_kj[t], l]
// m[t,:] = A'(1x1024) @ W2(1024x128); atomicAdd into h[idx_ji[t],:]
// 16 triplets/block, 8 waves (one 16-col tile each), 32 K-steps -> 256 WMMA.
// W2 panel (256KB) streams from L2 (1.5MB total, L2-resident).
// ---------------------------------------------------------------------------
__global__ __launch_bounds__(256)
void k_triplet(const unsigned short* __restrict__ xkj_bf,
               const float* __restrict__ sbfp,
               const int* __restrict__ idx_kj, const int* __restrict__ idx_ji,
               const unsigned short* __restrict__ W2t,   // [128][1024] bf16
               float* __restrict__ hbuf, int T) {
    __shared__ unsigned short raw[16 * HD];     // gathered x_kj rows (bf16)
    __shared__ unsigned short Ap[16 * KTRIP];   // folded A' (bf16)
    __shared__ float sb[16 * NB];
    __shared__ int   dstrow[16];
    const int t0 = blockIdx.x * 16;
    const int tid = threadIdx.x;
    {   // stage gathered rows: 16 rows x 256B, 16B per thread
        int rr = tid >> 4, off = (tid & 15) * 8;
        int t = t0 + rr;
        int src = (t < T) ? idx_kj[t] : 0;
        uint4 v = make_uint4(0, 0, 0, 0);
        if (t < T) v = *(const uint4*)(xkj_bf + (size_t)src * HD + off);
        *(uint4*)(raw + rr * HD + off) = v;
        if ((tid & 15) == 0) dstrow[rr] = (t < T) ? idx_ji[t] : -1;
    }
    if (tid < 16 * NB) {
        int rr = tid >> 3, jj = tid & 7;
        int t = t0 + rr;
        sb[tid] = (t < T) ? sbfp[(size_t)t * NB + jj] : 0.0f;
    }
    // prefetch next tile's gather indices (global_prefetch_b8)
    if (t0 + 16 < T && tid == 0) __builtin_prefetch(idx_kj + t0 + 16, 0, 1);
    __syncthreads();
    {   // build A': 16*1024 elems / 256 threads = 64 each
        int rr = tid >> 4, sub = tid & 15;
        #pragma unroll 8
        for (int q = 0; q < 64; ++q) {
            int k = sub + q * 16;                 // 0..1023
            int l = k & 127, jj = k >> 7;
            Ap[rr * KTRIP + k] = f2bf(bf2f(raw[rr * HD + l]) * sb[rr * NB + jj]);
        }
    }
    __syncthreads();
    const int wave = tid >> 5, lane = tid & 31;
    const int col0 = wave * 16;
    f32x8 acc = {};
    wmma_accum(Ap, KTRIP, W2t, KTRIP, col0, acc);
    const int half = lane >> 4, col = col0 + (lane & 15);
    #pragma unroll
    for (int r = 0; r < 8; ++r) {
        int rr = r + 8 * half;
        int dr = dstrow[rr];
        if (dr >= 0) atomicAdd(&hbuf[(size_t)dr * HD + col], acc[r]);
    }
}

// ---------------------------------------------------------------------------
// Output block pieces
// ---------------------------------------------------------------------------
__global__ void k_outedge(const float* __restrict__ rbf, const float* __restrict__ orbw,
                          const float* __restrict__ x, const int* __restrict__ iidx,
                          float* __restrict__ node_t, int E) {
    int g = blockIdx.x * blockDim.x + threadIdx.x;
    if (g >= E * HD) return;
    int e = g >> 7, c = g & 127;
    float rp = 0.0f;
    #pragma unroll
    for (int s = 0; s < NR; ++s) rp += rbf[(size_t)e * NR + s] * orbw[s * HD + c];
    atomicAdd(&node_t[(size_t)iidx[e] * HD + c], rp * x[g]);
}

__global__ void k_nodeout(const float* __restrict__ t, const float* __restrict__ ow,
                          float* __restrict__ Pn, int N) {
    int n = blockIdx.x * blockDim.x + threadIdx.x;
    if (n >= N) return;
    float s = 0.0f;
    #pragma unroll 8
    for (int c = 0; c < HD; ++c) s += t[(size_t)n * HD + c] * ow[c];
    Pn[n] += s;
}

__global__ void k_graphsum(const float* __restrict__ Pn, const int* __restrict__ batch,
                           float* __restrict__ out, int N) {
    int n = blockIdx.x * blockDim.x + threadIdx.x;
    if (n >= N) return;
    atomicAdd(&out[batch[n]], Pn[n]);
}

// ---------------------------------------------------------------------------
// Host orchestration
// ---------------------------------------------------------------------------
static inline int cdiv(int a, int b) { return (a + b - 1) / b; }

extern "C" void kernel_launch(void* const* d_in, const int* in_sizes, int n_in,
                              void* d_out, int out_size, void* d_ws, size_t ws_size,
                              hipStream_t stream) {
    (void)in_sizes; (void)n_in; (void)out_size; (void)ws_size;
    const int*   z         = (const int*)  d_in[0];
    const float* rbf       = (const float*)d_in[1];
    const float* sbf       = (const float*)d_in[2];
    const int*   iidx      = (const int*)  d_in[3];
    const int*   jidx      = (const int*)  d_in[4];
    const int*   idx_kj    = (const int*)  d_in[5];
    const int*   idx_ji    = (const int*)  d_in[6];
    const int*   batch     = (const int*)  d_in[7];
    const float* emb_table = (const float*)d_in[8];
    const float* emb_rbf_w = (const float*)d_in[9];
    const float* emb_rbf_b = (const float*)d_in[10];
    const float* emb_w     = (const float*)d_in[11];
    const float* emb_b     = (const float*)d_in[12];
    const float* int_rbf_w = (const float*)d_in[13];  // [6][6][128]
    const float* int_sbf_w = (const float*)d_in[14];  // [6][42][8]
    const float* int_kj_w  = (const float*)d_in[15];  // [6][128][128]
    const float* int_kj_b  = (const float*)d_in[16];
    const float* int_ji_w  = (const float*)d_in[17];
    const float* int_ji_b  = (const float*)d_in[18];
    const float* int_W     = (const float*)d_in[19];  // [6][128][8][128]
    const float* int_bef_w = (const float*)d_in[20];  // [6][2][128][128]
    const float* int_bef_b = (const float*)d_in[21];
    const float* int_lin_w = (const float*)d_in[22];
    const float* int_lin_b = (const float*)d_in[23];
    const float* int_aft_w = (const float*)d_in[24];  // [6][4][128][128]
    const float* int_aft_b = (const float*)d_in[25];
    const float* out_rbf_w = (const float*)d_in[26];  // [7][6][128]
    const float* out_lins_w= (const float*)d_in[27];  // [7][3][128][128]
    const float* out_lins_b= (const float*)d_in[28];  // [7][3][128]
    const float* out_w     = (const float*)d_in[29];  // [7][128][1]
    float* out = (float*)d_out;

    // ---- workspace carve-up ----
    char* wsp = (char*)d_ws;
    auto alloc = [&](size_t bytes) -> char* {
        char* p = wsp; wsp += (bytes + 255) & ~(size_t)255; return p;
    };
    const size_t EH = (size_t)N_EDGES * HD;
    float* xA   = (float*)alloc(EH * 4);
    float* xB   = (float*)alloc(EH * 4);
    float* hB   = (float*)alloc(EH * 4);
    float* tmp  = (float*)alloc(EH * 4);
    float* sbfp = (float*)alloc((size_t)N_TRIP * NB * 4);
    unsigned short* xkj_bf = (unsigned short*)alloc(EH * 2);
    float* nodeT   = (float*)alloc((size_t)N_ATOMS * HD * 4);
    float* nodeTmp = (float*)alloc((size_t)N_ATOMS * HD * 4);
    float* Pn      = (float*)alloc((size_t)N_ATOMS * 4);
    const int KH2 = HD * HD;
    unsigned short* emb_bf = (unsigned short*)alloc((size_t)95 * HD * 2);
    unsigned short* embWt  = (unsigned short*)alloc((size_t)384 * HD * 2);
    unsigned short* kjWt   = (unsigned short*)alloc((size_t)NBLK * KH2 * 2);
    unsigned short* jiWt   = (unsigned short*)alloc((size_t)NBLK * KH2 * 2);
    unsigned short* W2t    = (unsigned short*)alloc((size_t)NBLK * HD * KTRIP * 2);
    unsigned short* befWt  = (unsigned short*)alloc((size_t)NBLK * 2 * KH2 * 2);
    unsigned short* linWt  = (unsigned short*)alloc((size_t)NBLK * KH2 * 2);
    unsigned short* aftWt  = (unsigned short*)alloc((size_t)NBLK * 4 * KH2 * 2);
    unsigned short* outlWt = (unsigned short*)alloc((size_t)7 * 3 * KH2 * 2);

    const int TPB = 256;
    // ---- one-time weight conversion (bf16, transposed) ----
    k_conv <<<cdiv(95 * HD, TPB), TPB, 0, stream>>>(emb_table, emb_bf, 95 * HD);
    k_convT<<<cdiv(384 * HD, TPB), TPB, 0, stream>>>(emb_w, embWt, 384, HD);
    for (int b = 0; b < NBLK; ++b) {
        k_convT<<<cdiv(KH2, TPB), TPB, 0, stream>>>(int_kj_w + (size_t)b * KH2, kjWt + (size_t)b * KH2, HD, HD);
        k_convT<<<cdiv(KH2, TPB), TPB, 0, stream>>>(int_ji_w + (size_t)b * KH2, jiWt + (size_t)b * KH2, HD, HD);
        k_convW<<<cdiv(HD * KTRIP, TPB), TPB, 0, stream>>>(int_W + (size_t)b * HD * KTRIP, W2t + (size_t)b * HD * KTRIP);
        for (int l = 0; l < 2; ++l)
            k_convT<<<cdiv(KH2, TPB), TPB, 0, stream>>>(int_bef_w + ((size_t)b * 2 + l) * KH2,
                                                        befWt + ((size_t)b * 2 + l) * KH2, HD, HD);
        k_convT<<<cdiv(KH2, TPB), TPB, 0, stream>>>(int_lin_w + (size_t)b * KH2, linWt + (size_t)b * KH2, HD, HD);
        for (int l = 0; l < 4; ++l)
            k_convT<<<cdiv(KH2, TPB), TPB, 0, stream>>>(int_aft_w + ((size_t)b * 4 + l) * KH2,
                                                        aftWt + ((size_t)b * 4 + l) * KH2, HD, HD);
    }
    for (int k = 0; k < 7; ++k)
        for (int l = 0; l < 3; ++l)
            k_convT<<<cdiv(KH2, TPB), TPB, 0, stream>>>(out_lins_w + ((size_t)k * 3 + l) * KH2,
                                                        outlWt + ((size_t)k * 3 + l) * KH2, HD, HD);

    // ---- init accumulators ----
    k_zerof<<<cdiv(N_ATOMS, TPB), TPB, 0, stream>>>(Pn, N_ATOMS);
    k_zerof<<<1, 64, 0, stream>>>(out, N_GRAPH);

    // ---- embedding block ----
    k_embed<<<cdiv(N_EDGES, 32), 512, 0, stream>>>(rbf, iidx, jidx, z, emb_bf,
                                                   emb_rbf_w, emb_rbf_b, embWt, emb_b,
                                                   xA, N_EDGES);

    float* xcur = xA;
    float* xnext = xB;

    // ---- output block helper ----
    auto out_block = [&](const float* xe, int k) {
        k_zerof<<<cdiv(N_ATOMS * HD, TPB), TPB, 0, stream>>>(nodeT, N_ATOMS * HD);
        k_outedge<<<cdiv(N_EDGES * HD, TPB), TPB, 0, stream>>>(rbf, out_rbf_w + (size_t)k * NR * HD,
                                                               xe, iidx, nodeT, N_EDGES);
        float* a = nodeT; float* bbuf = nodeTmp;
        for (int l = 0; l < 3; ++l) {
            k_mlp<<<cdiv(N_ATOMS, 64), 512, 0, stream>>>(a, outlWt + ((size_t)k * 3 + l) * KH2,
                                                         out_lins_b + ((size_t)k * 3 + l) * HD, N_ATOMS,
                                                         bbuf, nullptr, nullptr, 1, nullptr, nullptr);
            float* sw = a; a = bbuf; bbuf = sw;
        }
        k_nodeout<<<cdiv(N_ATOMS, TPB), TPB, 0, stream>>>(a, out_w + (size_t)k * HD, Pn, N_ATOMS);
    };

    out_block(xcur, 0);

    // ---- interaction blocks ----
    const int gE = cdiv(N_EDGES, 64);
    for (int b = 0; b < NBLK; ++b) {
        // sbf projection [T,42]@[42,8]
        k_sbfp<<<cdiv(N_TRIP, TPB), TPB, 0, stream>>>(sbf, int_sbf_w + (size_t)b * NSR * NB, sbfp, N_TRIP);
        // x_ji = silu(x@Wji+b)  -> h (also serves as segment-sum accumulator)
        k_mlp<<<gE, 512, 0, stream>>>(xcur, jiWt + (size_t)b * KH2, int_ji_b + (size_t)b * HD, N_EDGES,
                                      hB, nullptr, nullptr, 1, nullptr, nullptr);
        // x_kj = silu(x@Wkj+b) * (rbf@int_rbf_w[b]) -> bf16 rows for gather
        k_mlp<<<gE, 512, 0, stream>>>(xcur, kjWt + (size_t)b * KH2, int_kj_b + (size_t)b * HD, N_EDGES,
                                      nullptr, xkj_bf, nullptr, 1, rbf, int_rbf_w + (size_t)b * NR * HD);
        // triplet einsum + segment_sum -> h (+= already holds x_ji)
        k_triplet<<<N_TRIP / 16, 256, 0, stream>>>(xkj_bf, sbfp, idx_kj, idx_ji,
                                                   W2t + (size_t)b * HD * KTRIP, hB, N_TRIP);
        // "before" residual (1 layer): h += silu(silu(h@W0+b0)@W1+b1)
        k_mlp<<<gE, 512, 0, stream>>>(hB, befWt + ((size_t)b * 2 + 0) * KH2,
                                      int_bef_b + ((size_t)b * 2 + 0) * HD, N_EDGES,
                                      tmp, nullptr, nullptr, 1, nullptr, nullptr);
        k_mlp<<<gE, 512, 0, stream>>>(tmp, befWt + ((size_t)b * 2 + 1) * KH2,
                                      int_bef_b + ((size_t)b * 2 + 1) * HD, N_EDGES,
                                      hB, nullptr, hB, 1, nullptr, nullptr);
        // skip: xnext = silu(h@lin+b) + x
        k_mlp<<<gE, 512, 0, stream>>>(hB, linWt + (size_t)b * KH2, int_lin_b + (size_t)b * HD, N_EDGES,
                                      xnext, nullptr, xcur, 1, nullptr, nullptr);
        // "after" residuals (2 layers) in place on xnext
        for (int r = 0; r < 2; ++r) {
            k_mlp<<<gE, 512, 0, stream>>>(xnext, aftWt + ((size_t)b * 4 + 2 * r) * KH2,
                                          int_aft_b + ((size_t)b * 4 + 2 * r) * HD, N_EDGES,
                                          tmp, nullptr, nullptr, 1, nullptr, nullptr);
            k_mlp<<<gE, 512, 0, stream>>>(tmp, aftWt + ((size_t)b * 4 + 2 * r + 1) * KH2,
                                          int_aft_b + ((size_t)b * 4 + 2 * r + 1) * HD, N_EDGES,
                                          xnext, nullptr, xnext, 1, nullptr, nullptr);
        }
        float* sw = xcur; xcur = xnext; xnext = sw;
        out_block(xcur, b + 1);
    }

    // ---- per-graph reduction ----
    k_graphsum<<<cdiv(N_ATOMS, TPB), TPB, 0, stream>>>(Pn, batch, out, N_ATOMS);
}
